// CausalSelfAttention_15436112462076
// MI455X (gfx1250) — matrix operand order repaired and checked by reading, not compiled, not run on hardware
//
#include <hip/hip_runtime.h>

// ---------- types ----------
typedef __attribute__((ext_vector_type(16))) __bf16          v16bf;
typedef __attribute__((ext_vector_type(8)))  float           v8f;
typedef __attribute__((ext_vector_type(4)))  float           f32x4;
typedef __attribute__((ext_vector_type(4)))  unsigned int    u32x4;
typedef __attribute__((ext_vector_type(4)))  unsigned short  u16x4;

union Frag { v16bf v; u32x4 q[2]; };

#define WMMA_BF16(a, b, c) \
  __builtin_amdgcn_wmma_f32_16x16x32_bf16(false, (a), false, (b), (short)0, (c), false, false)

// ---- CDNA5 async copy global -> LDS (ASYNCcnt-tracked DMA, no VGPR data) ----
__device__ __forceinline__ void async_b128(unsigned lds_off, const void* g) {
  asm volatile("global_load_async_to_lds_b128 %0, %1, off"
               :: "v"(lds_off), "v"((unsigned long long)(size_t)g) : "memory");
}
__device__ __forceinline__ void wait_async0() {
  asm volatile("s_wait_asynccnt 0" ::: "memory");
}
// LDS byte offset of a __shared__ object (generic addr low 32 bits = LDS offset)
#define LDS_OFF(p) ((unsigned)(size_t)(p))

__device__ __forceinline__ unsigned short f2bf(float f) {
  union { float f; unsigned u; } c; c.f = f;
  unsigned u = c.u;
  u += 0x7fffu + ((u >> 16) & 1u);   // round-to-nearest-even
  return (unsigned short)(u >> 16);
}

__device__ __forceinline__ float rmax16(float v) {
  v = fmaxf(v, __shfl_xor(v, 1, 32));
  v = fmaxf(v, __shfl_xor(v, 2, 32));
  v = fmaxf(v, __shfl_xor(v, 4, 32));
  v = fmaxf(v, __shfl_xor(v, 8, 32));
  return v;
}
__device__ __forceinline__ float rsum16(float v) {
  v += __shfl_xor(v, 1, 32);
  v += __shfl_xor(v, 2, 32);
  v += __shfl_xor(v, 4, 32);
  v += __shfl_xor(v, 8, 32);
  return v;
}

// ---------- fp32 -> bf16 convert (4 elems / thread) ----------
__global__ __launch_bounds__(256) void cvt_kernel(const float* __restrict__ in,
                                                  unsigned short* __restrict__ out, int n) {
  int i = (blockIdx.x * 256 + threadIdx.x) * 4;
  if (i >= n) return;
  f32x4 x = *(const f32x4*)(in + i);
  u16x4 y;
  y.x = f2bf(x.x); y.y = f2bf(x.y); y.z = f2bf(x.z); y.w = f2bf(x.w);
  *(u16x4*)(out + i) = y;
}

// ---------- pack W[K,N] fp32 -> per-lane B-fragment bf16 layout ----------
// tileId = kt*(N/16)+nt ; per tile: 32 lanes x 16 halves contiguous.
// lane<16 : column n = nt*16+lane,    K = kt*32 + 0..15
// lane>=16: column n = nt*16+lane-16, K = kt*32 + 16..31
__global__ __launch_bounds__(256) void pack_kernel(const float* __restrict__ W,
                                                   unsigned short* __restrict__ out,
                                                   int K, int N) {
  int t = blockIdx.x * 256 + threadIdx.x;
  int ntiles = N >> 4;
  int total = (K >> 5) * ntiles * 32;
  if (t >= total) return;
  int lane = t & 31;
  int tile = t >> 5;
  int kt = tile / ntiles, nt = tile % ntiles;
  int n = nt * 16 + (lane & 15);
  int kbase = kt * 32 + (lane >> 4) * 16;
  union { unsigned short us[16]; u32x4 q[2]; } tmp;
#pragma unroll
  for (int h = 0; h < 16; ++h)
    tmp.us[h] = f2bf(W[(size_t)(kbase + h) * N + n]);
  u32x4* dst = (u32x4*)(out + ((size_t)tile * 32 + lane) * 16);
  dst[0] = tmp.q[0];
  dst[1] = tmp.q[1];
}

// ---------- GEMM: C[M,N] = A(bf16)[M,K] * Wp(packed bf16) + bias ----------
// Block tile 128x64: 8 waves, each wave -> 16x64 (4 WMMA accumulators, A reused 4x).
// B tiles double-buffered in LDS via global_load_async_to_lds_b128 (ASYNCcnt DMA),
// one contiguous 4KB region per K-step = one b128 async per thread.
// grid = (N/64, M/128).
template <bool OUT_F32>
__global__ __launch_bounds__(256) void gemm_kernel(const unsigned short* __restrict__ A,
                                                   const unsigned short* __restrict__ Wp,
                                                   const float* __restrict__ bias,
                                                   float* __restrict__ Cf,
                                                   unsigned short* __restrict__ Cb,
                                                   int M, int N, int K) {
  __shared__ unsigned short Bl[2][2048];    // 2 x 4KB double buffer (4 B tiles)

  const int tid  = threadIdx.x;
  const int lane = tid & 31;
  const int wave = tid >> 5;
  const int lmod = lane & 15, lhi = lane >> 4;
  const int m0 = blockIdx.y * 128 + wave * 16;
  const int n0 = blockIdx.x * 64;
  const int ntiles = N >> 4;
  const int KT = K >> 5;

  const unsigned short* arow = A + (size_t)(m0 + lmod) * K + lhi * 16;
  const unsigned short* bsrc = Wp + (size_t)blockIdx.x * 4 * 512;   // + kt*ntiles*512
  const unsigned blbase = LDS_OFF(&Bl[0][0]);

  // prologue: DMA B tiles for kt=0 into buffer 0
  async_b128(blbase + tid * 16, bsrc + tid * 8);

  v8f acc0 = {}, acc1 = {}, acc2 = {}, acc3 = {};
  for (int kt = 0; kt < KT; ++kt) {
    wait_async0();
    __syncthreads();      // buf[cur] landed for all waves; buf[cur^1] free
    const int cur = kt & 1;
    if (kt + 1 < KT)      // overlap next DMA with this step's WMMAs
      async_b128(blbase + ((cur ^ 1) * 2048 + tid * 8) * 2,
                 bsrc + (size_t)(kt + 1) * ntiles * 512 + tid * 8);

    Frag af;
    const unsigned short* ap = arow + kt * 32;  // k = kt*32 + lhi*16 + {0..15}
    af.q[0] = *(const u32x4*)(ap);
    af.q[1] = *(const u32x4*)(ap + 8);

    const unsigned short* bb = &Bl[cur][lane * 16];
    Frag b0, b1, b2, b3;
    b0.q[0] = *(const u32x4*)(bb);        b0.q[1] = *(const u32x4*)(bb + 8);
    b1.q[0] = *(const u32x4*)(bb + 512);  b1.q[1] = *(const u32x4*)(bb + 520);
    b2.q[0] = *(const u32x4*)(bb + 1024); b2.q[1] = *(const u32x4*)(bb + 1032);
    b3.q[0] = *(const u32x4*)(bb + 1536); b3.q[1] = *(const u32x4*)(bb + 1544);
    acc0 = WMMA_BF16(af.v, b0.v, acc0);
    acc1 = WMMA_BF16(af.v, b1.v, acc1);
    acc2 = WMMA_BF16(af.v, b2.v, acc2);
    acc3 = WMMA_BF16(af.v, b3.v, acc3);
  }

  float bv0 = bias[n0 + lmod];
  float bv1 = bias[n0 + 16 + lmod];
  float bv2 = bias[n0 + 32 + lmod];
  float bv3 = bias[n0 + 48 + lmod];
#pragma unroll
  for (int r = 0; r < 8; ++r) {
    size_t idx = (size_t)(m0 + r + 8 * lhi) * N + n0 + lmod;
    float v0 = acc0[r] + bv0, v1 = acc1[r] + bv1;
    float v2 = acc2[r] + bv2, v3 = acc3[r] + bv3;
    if (OUT_F32) {
      Cf[idx] = v0; Cf[idx + 16] = v1; Cf[idx + 32] = v2; Cf[idx + 48] = v3;
    } else {
      Cb[idx] = f2bf(v0); Cb[idx + 16] = f2bf(v1);
      Cb[idx + 32] = f2bf(v2); Cb[idx + 48] = f2bf(v3);
    }
  }
}

// ---------- flash attention with the reference's INVERTED causal mask ----------
// grid = (S/128, B*H); 8 waves/block, each wave owns 16 query rows.
// Mask: score*0.125 + (j <= i ? -1e9 : 0)  (tril mask includes diagonal).
// K tile staged via async DMA to LDS; V transposed in-register (overlaps DMA).
__global__ __launch_bounds__(256) void attn_kernel(const unsigned short* __restrict__ Qp,
                                                   const unsigned short* __restrict__ Kp,
                                                   const unsigned short* __restrict__ Vp,
                                                   unsigned short* __restrict__ Op,
                                                   int S, int D) {
  __shared__ unsigned short Kl[32 * 64];    // K tile, [j][d] bf16
  __shared__ unsigned short Vt[64 * 32];    // V tile transposed, [d][j] bf16
  __shared__ unsigned short Pl[8][16 * 32]; // per-wave P bounce buffer

  const int lane = threadIdx.x & 31;
  const int wave = threadIdx.x >> 5;
  const int lmod = lane & 15, lhi = lane >> 4;
  const int b = blockIdx.y >> 4, h = blockIdx.y & 15;
  const int q0 = blockIdx.x * 128 + wave * 16;
  const size_t base = ((size_t)b * S) * D + h * 64;
  const unsigned klbase = LDS_OFF(&Kl[0]);

  // Q fragments: A-layout, two 32-deep chunks over head dim 64
  Frag aq0, aq1;
  {
    const unsigned short* qrow = Qp + base + (size_t)(q0 + lmod) * D + lhi * 16;
    aq0.q[0] = *(const u32x4*)(qrow);
    aq0.q[1] = *(const u32x4*)(qrow + 8);
    aq1.q[0] = *(const u32x4*)(qrow + 32);
    aq1.q[1] = *(const u32x4*)(qrow + 40);
  }

  float ms[8], ls[8];
  v8f o0 = {}, o1 = {}, o2 = {}, o3 = {};
#pragma unroll
  for (int r = 0; r < 8; ++r) { ms[r] = -1e30f; ls[r] = 0.f; }

  for (int j0 = 0; j0 < S; j0 += 32) {
    __syncthreads();    // previous step's reads of Kl/Vt complete
    { // cooperative staging: 256 threads x 8 bf16 elems each
      int e = threadIdx.x * 8;
      int j = e >> 6, d = e & 63;
      const size_t goff = base + (size_t)(j0 + j) * D + d;
      async_b128(klbase + e * 2, &Kp[goff]);          // DMA K row chunk -> LDS
      u32x4 vv = *(const u32x4*)&Vp[goff];            // V: load + transpose
      unsigned v0 = vv.x, v1 = vv.y, v2 = vv.z, v3 = vv.w;
      Vt[(d + 0) * 32 + j] = (unsigned short)v0;
      Vt[(d + 1) * 32 + j] = (unsigned short)(v0 >> 16);
      Vt[(d + 2) * 32 + j] = (unsigned short)v1;
      Vt[(d + 3) * 32 + j] = (unsigned short)(v1 >> 16);
      Vt[(d + 4) * 32 + j] = (unsigned short)v2;
      Vt[(d + 5) * 32 + j] = (unsigned short)(v2 >> 16);
      Vt[(d + 6) * 32 + j] = (unsigned short)v3;
      Vt[(d + 7) * 32 + j] = (unsigned short)(v3 >> 16);
      if (j0 + 32 < S) { // prefetch next tile -> global_prefetch_b8
        __builtin_prefetch(&Kp[goff + (size_t)32 * D], 0, 0);
        __builtin_prefetch(&Vp[goff + (size_t)32 * D], 0, 0);
      }
      wait_async0();    // my K DMA landed
    }
    __syncthreads();    // everyone's K DMA + V stores visible

    // scores: S[16,32] = Q[16,64] @ K^T[64,32]   (two 16-wide j tiles)
    v8f s0 = {}, s1 = {};
#pragma unroll
    for (int jt = 0; jt < 2; ++jt) {
      Frag bk0, bk1;
      const unsigned short* p0 = &Kl[(jt * 16 + lmod) * 64 + lhi * 16];        // d chunk 0
      bk0.q[0] = *(const u32x4*)(p0);
      bk0.q[1] = *(const u32x4*)(p0 + 8);
      const unsigned short* p1 = &Kl[(jt * 16 + lmod) * 64 + 32 + lhi * 16];   // d chunk 1
      bk1.q[0] = *(const u32x4*)(p1);
      bk1.q[1] = *(const u32x4*)(p1 + 8);
      if (jt == 0) { s0 = WMMA_BF16(aq0.v, bk0.v, s0); s0 = WMMA_BF16(aq1.v, bk1.v, s0); }
      else         { s1 = WMMA_BF16(aq0.v, bk0.v, s1); s1 = WMMA_BF16(aq1.v, bk1.v, s1); }
    }

    // mask + online softmax (rows live in lane-groups of 16, vgpr r -> row r+8*lhi)
    const int jA = j0 + lmod, jB = j0 + 16 + lmod;
    float mnew[8];
#pragma unroll
    for (int r = 0; r < 8; ++r) {
      int i = q0 + r + 8 * lhi;
      float a = s0[r] * 0.125f + (jA <= i ? -1e9f : 0.f);
      float c = s1[r] * 0.125f + (jB <= i ? -1e9f : 0.f);
      s0[r] = a; s1[r] = c;
      mnew[r] = fmaxf(ms[r], rmax16(fmaxf(a, c)));
    }
    unsigned short* pw = Pl[wave];
#pragma unroll
    for (int r = 0; r < 8; ++r) {
      float sc = __expf(ms[r] - mnew[r]);
      ms[r] = mnew[r];
      ls[r] *= sc;
      o0[r] *= sc; o1[r] *= sc; o2[r] *= sc; o3[r] *= sc;
      float p0 = __expf(s0[r] - mnew[r]);
      float p1 = __expf(s1[r] - mnew[r]);
      ls[r] += rsum16(p0 + p1);
      int row = r + 8 * lhi;
      pw[row * 32 + lmod]      = f2bf(p0);
      pw[row * 32 + 16 + lmod] = f2bf(p1);
    }

    // reload P as A-fragment (wave-private LDS, DS ops in order within wave)
    Frag ap;
    const unsigned short* pp = pw + lmod * 32 + lhi * 16;
    ap.q[0] = *(const u32x4*)(pp);
    ap.q[1] = *(const u32x4*)(pp + 8);

    // O[16,64] += P[16,32] @ V[32,64]  (4 n-tiles; B frags from transposed Vt)
#pragma unroll
    for (int nt = 0; nt < 4; ++nt) {
      Frag bvf;
      const unsigned short* vp = &Vt[(nt * 16 + lmod) * 32 + lhi * 16];
      bvf.q[0] = *(const u32x4*)(vp);
      bvf.q[1] = *(const u32x4*)(vp + 8);
      if      (nt == 0) o0 = WMMA_BF16(ap.v, bvf.v, o0);
      else if (nt == 1) o1 = WMMA_BF16(ap.v, bvf.v, o1);
      else if (nt == 2) o2 = WMMA_BF16(ap.v, bvf.v, o2);
      else              o3 = WMMA_BF16(ap.v, bvf.v, o3);
    }
  }

  // epilogue: O /= l, write bf16 into attn buffer [B,S,D]
#pragma unroll
  for (int r = 0; r < 8; ++r) {
    float inv = 1.f / ls[r];
    size_t rowo = base + (size_t)(q0 + r + 8 * lhi) * D + lmod;
    Op[rowo]      = f2bf(o0[r] * inv);
    Op[rowo + 16] = f2bf(o1[r] * inv);
    Op[rowo + 32] = f2bf(o2[r] * inv);
    Op[rowo + 48] = f2bf(o3[r] * inv);
  }
}

// ---------- host ----------
extern "C" void kernel_launch(void* const* d_in, const int* in_sizes, int n_in,
                              void* d_out, int out_size, void* d_ws, size_t ws_size,
                              hipStream_t stream) {
  const int B = 2, S = 2048, D = 1024;
  const int M = B * S;                       // 4096
  const float* X  = (const float*)d_in[0];
  const float* Wq = (const float*)d_in[1];
  const float* bq = (const float*)d_in[2];
  const float* Wk = (const float*)d_in[3];
  const float* bk = (const float*)d_in[4];
  const float* Wv = (const float*)d_in[5];
  const float* bv = (const float*)d_in[6];
  const float* Wo = (const float*)d_in[7];
  const float* bo = (const float*)d_in[8];

  unsigned short* ws = (unsigned short*)d_ws;
  const size_t MD = (size_t)M * D;           // 4M elements
  unsigned short* Xb  = ws;                  // bf16 activations
  unsigned short* Qb  = ws + MD;
  unsigned short* Kb  = ws + 2 * MD;
  unsigned short* Vb  = ws + 3 * MD;
  unsigned short* Ab  = ws + 4 * MD;         // attention output (bf16)
  unsigned short* Wqp = ws + 5 * MD;         // packed weights, 1M elems each
  unsigned short* Wkp = Wqp + (size_t)D * D;
  unsigned short* Wvp = Wkp + (size_t)D * D;
  unsigned short* Wop = Wvp + (size_t)D * D;

  cvt_kernel<<<(int)(MD / 4 / 256), 256, 0, stream>>>(X, Xb, (int)MD);

  const int packT = (D / 32) * (D / 16) * 32;    // 65536
  pack_kernel<<<packT / 256, 256, 0, stream>>>(Wq, Wqp, D, D);
  pack_kernel<<<packT / 256, 256, 0, stream>>>(Wk, Wkp, D, D);
  pack_kernel<<<packT / 256, 256, 0, stream>>>(Wv, Wvp, D, D);
  pack_kernel<<<packT / 256, 256, 0, stream>>>(Wo, Wop, D, D);

  dim3 gg(D / 64, M / 128);                  // (16, 32)
  gemm_kernel<false><<<gg, 256, 0, stream>>>(Xb, Wqp, bq, nullptr, Qb, M, D, D);
  gemm_kernel<false><<<gg, 256, 0, stream>>>(Xb, Wkp, bk, nullptr, Kb, M, D, D);
  gemm_kernel<false><<<gg, 256, 0, stream>>>(Xb, Wvp, bv, nullptr, Vb, M, D, D);

  attn_kernel<<<dim3(S / 128, B * 16), 256, 0, stream>>>(Qb, Kb, Vb, Ab, S, D);

  gemm_kernel<true><<<gg, 256, 0, stream>>>(Ab, Wop, bo, (float*)d_out, nullptr, M, D, D);
}
